// GATEncoder_90795608638298
// MI455X (gfx1250) — compile-verified, matrix operand order checked by reading, not run on hardware
//
#include <hip/hip_runtime.h>
#include <hip/hip_bf16.h>
#include <math.h>

// ---------------- problem constants (from reference) ----------------
constexpr int N_NODES = 20000;
constexpr int N_EDGES = 320000;
constexpr int BGR     = 512;
constexpr int D       = 256;   // hidden dim = H*C
constexpr int H       = 8;
constexpr int C       = 32;
constexpr int DE      = 64;
constexpr int L       = 6;
constexpr int P       = 512;
constexpr int FN      = 74;
constexpr int FE      = 12;
constexpr int EF      = N_EDGES + N_NODES;   // edges + self loops = 340000

#define DEV __device__ __forceinline__

// ---------------- small helpers ----------------
typedef __attribute__((ext_vector_type(16))) __bf16 v16bf;
typedef __attribute__((ext_vector_type(8)))  float  v8f;

struct alignas(16) U128 { unsigned int x, y, z, w; };

DEV unsigned short f2bf(float f) {
  unsigned u = __float_as_uint(f);
  unsigned r = u + 0x7fffu + ((u >> 16) & 1u);   // RNE
  return (unsigned short)(r >> 16);
}
DEV float bf2f(unsigned short h) { return __uint_as_float(((unsigned)h) << 16); }

DEV float gelu_f(float x) { return 0.5f * x * (1.0f + erff(x * 0.70710678118654752f)); }

// order-preserving encoding of float -> uint for atomicMax
DEV unsigned encOrd(float f) {
  unsigned u = __float_as_uint(f);
  return (u & 0x80000000u) ? ~u : (u | 0x80000000u);
}
DEV float decOrd(unsigned m) {
  return __uint_as_float((m & 0x80000000u) ? (m & 0x7FFFFFFFu) : ~m);
}

template<int NT>
DEV float block_sum(float v, float* sh) {
#pragma unroll
  for (int o = 16; o; o >>= 1) v += __shfl_xor(v, o);
  const int w = threadIdx.x >> 5;
  if ((threadIdx.x & 31) == 0) sh[w] = v;
  __syncthreads();
  float s = 0.0f;
  if (threadIdx.x < (NT / 32)) s = sh[threadIdx.x];
  if (threadIdx.x < 32) {
#pragma unroll
    for (int o = 16; o; o >>= 1) s += __shfl_xor(s, o);
    if (threadIdx.x == 0) sh[0] = s;
  }
  __syncthreads();
  s = sh[0];
  __syncthreads();
  return s;
}

// ---------------- WMMA bf16 GEMM: C[M,Nout] = A[M,K] * BT[Nout,K]^T + bias ----------------
// A row-major bf16 (ld=K); BT row-major bf16 (ld=K)  (i.e. B stored transposed).
// 16-bit A-matrix fragment layout (CDNA5 ISA 7.12.2): lane m=lane&15,
// lanes 0-15 hold K = k0+0..7 and k0+16..23 ; lanes 16-31 hold K = k0+8..15 and k0+24..31.
DEV v16bf load_frag(const unsigned short* __restrict__ base, int ld, int row, int k0) {
  const int lane = threadIdx.x & 31;
  const int m  = lane & 15;
  const int kb = (lane & 16) >> 1;               // 0 or 8
  const unsigned short* p = base + (size_t)(row + m) * (size_t)ld + (k0 + kb);
  union { U128 u[2]; v16bf v; } f;
  f.u[0] = *reinterpret_cast<const U128*>(p);        // K = k0+kb    .. +7  (16B)
  f.u[1] = *reinterpret_cast<const U128*>(p + 16);   // K = k0+16+kb .. +7  (16B)
  return f.v;
}

__global__ __launch_bounds__(256) void wmma_gemm_bf16_k(
    const unsigned short* __restrict__ A,
    const unsigned short* __restrict__ BT,
    const float* __restrict__ bias,
    float* __restrict__ Cf,                 // fp32 output (or null)
    unsigned short* __restrict__ Cbf,       // bf16 output, streamed NT (used when Cf==null)
    int M, int K, int Nout) {
  const int lane = threadIdx.x & 31;
  const int wave = threadIdx.x >> 5;
  const int row0 = blockIdx.x * 32 + (wave >> 2) * 16;   // 2 row-groups of 16
  const int col0 = blockIdx.y * 256 + (wave & 3) * 64;   // 4 col-groups of 64
  if (row0 >= M || col0 >= Nout) return;                 // wave-uniform guard: EXEC stays all-1s

  v8f acc[4] = {v8f{}, v8f{}, v8f{}, v8f{}};
  for (int k0 = 0; k0 < K; k0 += 32) {
    v16bf a = load_frag(A, K, row0, k0);
#pragma unroll
    for (int t = 0; t < 4; ++t) {
      v16bf b = load_frag(BT, K, col0 + t * 16, k0);
      acc[t] = __builtin_amdgcn_wmma_f32_16x16x32_bf16(
          /*neg_a=*/false, a, /*neg_b=*/false, b,
          /*c_mod=*/(short)0, acc[t], /*reuse_a=*/false, /*reuse_b=*/false);
    }
  }
  // C/D layout: lane n = lane&15 ; VGPR j -> row j (lanes 0-15) / j+8 (lanes 16-31)
  const int n    = lane & 15;
  const int mofs = (lane & 16) >> 1;
#pragma unroll
  for (int t = 0; t < 4; ++t) {
    const int col = col0 + t * 16 + n;
    const float bv = bias ? bias[col] : 0.0f;
#pragma unroll
    for (int j = 0; j < 8; ++j) {
      const float v = acc[t][j] + bv;
      const size_t idx = (size_t)(row0 + mofs + j) * (size_t)Nout + col;
      if (Cf) Cf[idx] = v;
      else    __builtin_nontemporal_store(f2bf(v), &Cbf[idx]);  // stream: don't pollute L2
    }
  }
}

// ---------------- elementwise / prep kernels ----------------
__global__ void fill_u32_k(unsigned* __restrict__ p, unsigned v, size_t n) {
  size_t i = (size_t)blockIdx.x * blockDim.x + threadIdx.x;
  if (i < n) p[i] = v;
}

__global__ void f32_to_bf16_k(const float* __restrict__ s, unsigned short* __restrict__ d, size_t n) {
  size_t i = (size_t)blockIdx.x * blockDim.x + threadIdx.x;
  if (i < n) d[i] = f2bf(s[i]);
}

// W[K,Nout] fp32 -> WT[Nout,K] bf16
__global__ void transpose_bf16_k(const float* __restrict__ W, unsigned short* __restrict__ WT,
                                 int K, int Nout) {
  int i = blockIdx.x * 256 + threadIdx.x;
  if (i >= K * Nout) return;
  int n = i / K, k = i - n * K;
  WT[i] = f2bf(W[(size_t)k * Nout + n]);
}

// h = LN(GELU(x @ node_w + node_b)) ; one block per node
__global__ __launch_bounds__(256) void node_embed_k(
    const float* __restrict__ x, const float* __restrict__ w, const float* __restrict__ b,
    const float* __restrict__ g, const float* __restrict__ beta,
    float* __restrict__ h, unsigned short* __restrict__ hbf) {
  __shared__ float xs[FN];
  __shared__ float sh[8];
  const int nd = blockIdx.x, t = threadIdx.x;
  if (t < FN) xs[t] = x[(size_t)nd * FN + t];
  __syncthreads();
  float acc = b[t];
  for (int k = 0; k < FN; ++k) acc = fmaf(xs[k], w[k * D + t], acc);
  acc = gelu_f(acc);
  const float mu  = block_sum<256>(acc, sh) * (1.0f / D);
  const float dv  = acc - mu;
  const float var = block_sum<256>(dv * dv, sh) * (1.0f / D);
  const float v   = dv * rsqrtf(var + 1e-5f) * g[t] + beta[t];
  h[(size_t)nd * D + t]   = v;
  hbf[(size_t)nd * D + t] = f2bf(v);
}

// e = GELU(edge_attr @ edge_w + edge_b) ; one block (64 thr) per edge
__global__ __launch_bounds__(64) void edge_embed_k(
    const float* __restrict__ ea, const float* __restrict__ w, const float* __restrict__ b,
    float* __restrict__ ef) {
  __shared__ float s[FE];
  const int e = blockIdx.x, t = threadIdx.x;
  if (t < FE) s[t] = ea[(size_t)e * FE + t];
  __syncthreads();
  float acc = b[t];
  for (int k = 0; k < FE; ++k) acc = fmaf(s[k], w[k * DE + t], acc);
  ef[(size_t)e * DE + t] = gelu_f(acc);
}

// self-loop attr = mean of incoming edge attrs (accumulate into ef rows [E, E+N))
__global__ void loop_accum_k(const float* __restrict__ ef, const int* __restrict__ dst,
                             float* __restrict__ loopsum, float* __restrict__ cnt) {
  size_t i = (size_t)blockIdx.x * blockDim.x + threadIdx.x;
  if (i >= (size_t)N_EDGES * DE) return;
  int e = (int)(i >> 6), c = (int)(i & 63);
  int d = dst[e];
  atomicAdd(&loopsum[(size_t)d * DE + c], ef[i]);
  if (c == 0) atomicAdd(&cnt[d], 1.0f);
}

__global__ void loop_finish_k(float* __restrict__ loopsum, const float* __restrict__ cnt) {
  size_t i = (size_t)blockIdx.x * blockDim.x + threadIdx.x;
  if (i >= (size_t)N_NODES * DE) return;
  int n = (int)(i >> 6);
  loopsum[i] /= fmaxf(cnt[n], 1.0f);
}

// persistent waves, one edge per wave-iteration:
// m = leaky(xl[src]+xr[dst]+em) ; alpha_logit = sum_c m*att ; atomicMax per dst
// prefetches next iteration's gather rows (global_prefetch_b8) while reducing current
__global__ __launch_bounds__(256) void edge_alpha_k(
    const float* __restrict__ xl, const float* __restrict__ xr,
    const unsigned short* __restrict__ em,
    const int* __restrict__ src, const int* __restrict__ dst,
    const float* __restrict__ att,
    float* __restrict__ logit, unsigned* __restrict__ nmax) {
  const int lane = threadIdx.x & 31;
  const int wid  = blockIdx.x * 8 + (threadIdx.x >> 5);
  const int nw   = gridDim.x * 8;
  for (int gid = wid; gid < EF; gid += nw) {
    // software-pipeline: prefetch next edge's node rows (1KB each; 1 line / 8 lanes)
    const int ng = gid + nw;
    if (ng < EF) {
      const int ns = (ng < N_EDGES) ? src[ng] : (ng - N_EDGES);
      const int nd = (ng < N_EDGES) ? dst[ng] : (ng - N_EDGES);
      __builtin_prefetch(xl + (size_t)ns * D + lane * 8, 0, 3);
      __builtin_prefetch(xr + (size_t)nd * D + lane * 8, 0, 3);
    }
    const int s = (gid < N_EDGES) ? src[gid] : (gid - N_EDGES);
    const int d = (gid < N_EDGES) ? dst[gid] : (gid - N_EDGES);
    const float* pl = xl + (size_t)s * D;
    const float* pr = xr + (size_t)d * D;
    const unsigned short* pe = em + (size_t)gid * D;
#pragma unroll
    for (int h = 0; h < H; ++h) {
      const int c = h * C + lane;
      float v = pl[c] + pr[c] + bf2f(__builtin_nontemporal_load(&pe[c]));  // em is stream-once
      v = (v > 0.0f) ? v : 0.2f * v;                 // leaky_relu(0.2)
      float p = v * att[h * C + lane];
#pragma unroll
      for (int o = 16; o; o >>= 1) p += __shfl_xor(p, o);
      if (lane == 0) {
        logit[(size_t)gid * H + h] = p;
        atomicMax(&nmax[(size_t)d * H + h], encOrd(p));
      }
    }
  }
}

// ex = exp(logit - max[dst]) (in place) ; atomicAdd to per-dst sum
__global__ void alpha_exp_k(float* __restrict__ lg, const unsigned* __restrict__ nmax,
                            float* __restrict__ nsum, const int* __restrict__ dst) {
  size_t i = (size_t)blockIdx.x * blockDim.x + threadIdx.x;
  if (i >= (size_t)EF * H) return;
  int gid = (int)(i >> 3), h = (int)(i & 7);
  int d = (gid < N_EDGES) ? dst[gid] : (gid - N_EDGES);
  float m = decOrd(nmax[(size_t)d * H + h]);
  float ex = expf(lg[i] - m);
  lg[i] = ex;
  atomicAdd(&nsum[(size_t)d * H + h], ex);
}

// out[dst] += (ex/sum) * xl[src] ; persistent blocks (256 thr = channels), grid-stride
// over edges with prefetch of the next edge's xl row.
__global__ __launch_bounds__(256) void aggregate_k(
    const float* __restrict__ ex, const float* __restrict__ nsum,
    const float* __restrict__ xl,
    const int* __restrict__ src, const int* __restrict__ dst,
    float* __restrict__ out) {
  const int ch = threadIdx.x;
  const int h  = ch >> 5;
  for (int gid = blockIdx.x; gid < EF; gid += gridDim.x) {
    const int ng = gid + gridDim.x;
    if (ng < EF && (ch & 31) == 0) {                  // 8 lanes cover the 1KB row
      const int ns = (ng < N_EDGES) ? src[ng] : (ng - N_EDGES);
      __builtin_prefetch(xl + (size_t)ns * D + ch, 0, 3);
    }
    const int s = (gid < N_EDGES) ? src[gid] : (gid - N_EDGES);
    const int d = (gid < N_EDGES) ? dst[gid] : (gid - N_EDGES);
    const float a = ex[(size_t)gid * H + h] / (nsum[(size_t)d * H + h] + 1e-16f);
    atomicAdd(&out[(size_t)d * D + ch], a * xl[(size_t)s * D + ch]);
  }
}

// h = h + GELU(LN(out + conv_b)) ; refresh bf16 copy
__global__ __launch_bounds__(256) void layer_post_k(
    const float* __restrict__ out, const float* __restrict__ cb,
    const float* __restrict__ g, const float* __restrict__ b,
    float* __restrict__ h, unsigned short* __restrict__ hbf) {
  __shared__ float sh[8];
  const int n = blockIdx.x, t = threadIdx.x;
  const float v0  = out[(size_t)n * D + t] + cb[t];
  const float mu  = block_sum<256>(v0, sh) * (1.0f / D);
  const float dv  = v0 - mu;
  const float var = block_sum<256>(dv * dv, sh) * (1.0f / D);
  const float ln  = dv * rsqrtf(var + 1e-5f) * g[t] + b[t];
  const float nh  = h[(size_t)n * D + t] + gelu_f(ln);
  h[(size_t)n * D + t]   = nh;
  hbf[(size_t)n * D + t] = f2bf(nh);
}

// pooling: xs[batch[n]] += h[n] ; gcnt[batch[n]] += 1
__global__ __launch_bounds__(256) void pool_accum_k(
    const float* __restrict__ h, const int* __restrict__ batch,
    float* __restrict__ xs, float* __restrict__ gcnt) {
  const int n = blockIdx.x, t = threadIdx.x;
  const int b = batch[n];
  atomicAdd(&xs[(size_t)b * D + t], h[(size_t)n * D + t]);
  if (t == 0) atomicAdd(&gcnt[b], 1.0f);
}

// xg = concat(mean, sum) [B, 2D] (fp32 + bf16)
__global__ __launch_bounds__(512) void build_xg_k(
    const float* __restrict__ xs, const float* __restrict__ gcnt,
    float* __restrict__ xg, unsigned short* __restrict__ xgbf) {
  const int b = blockIdx.x, t = threadIdx.x;
  const float cn = fmaxf(gcnt[b], 1.0f);
  const float v = (t < D) ? (xs[(size_t)b * D + t] / cn) : xs[(size_t)b * D + (t - D)];
  xg[(size_t)b * (2 * D) + t]   = v;
  xgbf[(size_t)b * (2 * D) + t] = f2bf(v);
}

// z = LN(GELU(z)) in place ; also bf16 copy
__global__ __launch_bounds__(256) void post_mlp1_k(
    float* __restrict__ z, const float* __restrict__ g, const float* __restrict__ b,
    unsigned short* __restrict__ zbf) {
  __shared__ float sh[8];
  const int r = blockIdx.x, t = threadIdx.x;
  const float v0  = gelu_f(z[(size_t)r * D + t]);
  const float mu  = block_sum<256>(v0, sh) * (1.0f / D);
  const float dv  = v0 - mu;
  const float var = block_sum<256>(dv * dv, sh) * (1.0f / D);
  const float v   = dv * rsqrtf(var + 1e-5f) * g[t] + b[t];
  z[(size_t)r * D + t]   = v;
  zbf[(size_t)r * D + t] = f2bf(v);
}

// out = LN(opre) over width P=512
__global__ __launch_bounds__(512) void final_ln_k(
    const float* __restrict__ opre, const float* __restrict__ g, const float* __restrict__ b,
    float* __restrict__ out) {
  __shared__ float sh[16];
  const int r = blockIdx.x, t = threadIdx.x;
  const float v0  = opre[(size_t)r * P + t];
  const float mu  = block_sum<512>(v0, sh) * (1.0f / P);
  const float dv  = v0 - mu;
  const float var = block_sum<512>(dv * dv, sh) * (1.0f / P);
  out[(size_t)r * P + t] = dv * rsqrtf(var + 1e-5f) * g[t] + b[t];
}

// ---------------- host-side orchestration ----------------
static inline unsigned cdivu(size_t a, size_t b) { return (unsigned)((a + b - 1) / b); }

extern "C" void kernel_launch(void* const* d_in, const int* in_sizes, int n_in,
                              void* d_out, int out_size, void* d_ws, size_t ws_size,
                              hipStream_t stream) {
  (void)in_sizes; (void)n_in; (void)out_size; (void)ws_size;

  const float* x        = (const float*)d_in[0];
  const float* eattr    = (const float*)d_in[1];
  const float* node_w   = (const float*)d_in[2];
  const float* node_b   = (const float*)d_in[3];
  const float* node_g   = (const float*)d_in[4];
  const float* node_be  = (const float*)d_in[5];
  const float* edge_w   = (const float*)d_in[6];
  const float* edge_b   = (const float*)d_in[7];
  const float* lin_l_w  = (const float*)d_in[8];
  const float* lin_l_b  = (const float*)d_in[9];
  const float* lin_r_w  = (const float*)d_in[10];
  const float* lin_r_b  = (const float*)d_in[11];
  const float* lin_e_w  = (const float*)d_in[12];
  const float* att_w    = (const float*)d_in[13];
  const float* conv_b   = (const float*)d_in[14];
  const float* ln_g     = (const float*)d_in[15];
  const float* ln_b     = (const float*)d_in[16];
  const float* p1_w     = (const float*)d_in[17];
  const float* p1_b     = (const float*)d_in[18];
  const float* pln1_g   = (const float*)d_in[19];
  const float* pln1_b   = (const float*)d_in[20];
  const float* p2_w     = (const float*)d_in[21];
  const float* p2_b     = (const float*)d_in[22];
  const float* pln2_g   = (const float*)d_in[23];
  const float* pln2_b   = (const float*)d_in[24];
  const int*   eidx     = (const int*)d_in[25];
  const int*   batch    = (const int*)d_in[26];
  const int* esrc = eidx;
  const int* edst = eidx + N_EDGES;

  // ---- workspace bump allocator ----
  char* wsp = (char*)d_ws;
  auto alloc = [&](size_t bytes) -> void* {
    void* p = wsp; wsp += (bytes + 255) & ~(size_t)255; return p;
  };
  float*          h      = (float*)alloc((size_t)N_NODES * D * 4);
  unsigned short* hbf    = (unsigned short*)alloc((size_t)N_NODES * D * 2);
  float*          ef32   = (float*)alloc((size_t)EF * DE * 4);       // rows [E,EF) = self-loop attrs
  unsigned short* efbf   = (unsigned short*)alloc((size_t)EF * DE * 2);
  float*          cnt    = (float*)alloc((size_t)N_NODES * 4);
  unsigned short* wTl    = (unsigned short*)alloc((size_t)D * D * 2);
  unsigned short* wTr    = (unsigned short*)alloc((size_t)D * D * 2);
  unsigned short* wTe    = (unsigned short*)alloc((size_t)D * DE * 2);
  float*          xl     = (float*)alloc((size_t)N_NODES * D * 4);
  float*          xr     = (float*)alloc((size_t)N_NODES * D * 4);
  unsigned short* embf   = (unsigned short*)alloc((size_t)EF * D * 2);
  float*          logit  = (float*)alloc((size_t)EF * H * 4);
  unsigned*       nmax   = (unsigned*)alloc((size_t)N_NODES * H * 4);
  float*          nsum   = (float*)alloc((size_t)N_NODES * H * 4);
  float*          outacc = (float*)alloc((size_t)N_NODES * D * 4);
  float*          xs     = (float*)alloc((size_t)BGR * D * 4);
  float*          gcnt   = (float*)alloc((size_t)BGR * 4);
  float*          xg     = (float*)alloc((size_t)BGR * 2 * D * 4);
  unsigned short* xgbf   = (unsigned short*)alloc((size_t)BGR * 2 * D * 2);
  unsigned short* p1T    = (unsigned short*)alloc((size_t)D * 2 * D * 2);   // [256,512]
  float*          z      = (float*)alloc((size_t)BGR * D * 4);
  unsigned short* zbf    = (unsigned short*)alloc((size_t)BGR * D * 2);
  unsigned short* p2T    = (unsigned short*)alloc((size_t)P * D * 2);       // [512,256]
  float*          opre   = (float*)alloc((size_t)BGR * P * 4);

  // ---- embeddings ----
  node_embed_k<<<N_NODES, 256, 0, stream>>>(x, node_w, node_b, node_g, node_be, h, hbf);
  edge_embed_k<<<N_EDGES, 64, 0, stream>>>(eattr, edge_w, edge_b, ef32);

  // ---- self loops (mean of incoming edge attrs) ----
  fill_u32_k<<<cdivu((size_t)N_NODES * DE, 256), 256, 0, stream>>>(
      (unsigned*)(ef32 + (size_t)N_EDGES * DE), 0u, (size_t)N_NODES * DE);
  fill_u32_k<<<cdivu((size_t)N_NODES, 256), 256, 0, stream>>>((unsigned*)cnt, 0u, (size_t)N_NODES);
  loop_accum_k<<<cdivu((size_t)N_EDGES * DE, 256), 256, 0, stream>>>(
      ef32, edst, ef32 + (size_t)N_EDGES * DE, cnt);
  loop_finish_k<<<cdivu((size_t)N_NODES * DE, 256), 256, 0, stream>>>(
      ef32 + (size_t)N_EDGES * DE, cnt);
  f32_to_bf16_k<<<cdivu((size_t)EF * DE, 256), 256, 0, stream>>>(ef32, efbf, (size_t)EF * DE);

  // ---- GAT layers ----
  for (int l = 0; l < L; ++l) {
    transpose_bf16_k<<<cdivu((size_t)D * D, 256), 256, 0, stream>>>(
        lin_l_w + (size_t)l * D * D, wTl, D, D);
    transpose_bf16_k<<<cdivu((size_t)D * D, 256), 256, 0, stream>>>(
        lin_r_w + (size_t)l * D * D, wTr, D, D);
    transpose_bf16_k<<<cdivu((size_t)DE * D, 256), 256, 0, stream>>>(
        lin_e_w + (size_t)l * DE * D, wTe, DE, D);

    dim3 gN(N_NODES / 32, 1);
    wmma_gemm_bf16_k<<<gN, 256, 0, stream>>>(hbf, wTl, lin_l_b + (size_t)l * D,
                                             xl, nullptr, N_NODES, D, D);
    wmma_gemm_bf16_k<<<gN, 256, 0, stream>>>(hbf, wTr, lin_r_b + (size_t)l * D,
                                             xr, nullptr, N_NODES, D, D);
    dim3 gE(EF / 32, 1);
    wmma_gemm_bf16_k<<<gE, 256, 0, stream>>>(efbf, wTe, nullptr,
                                             nullptr, embf, EF, DE, D);

    fill_u32_k<<<cdivu((size_t)N_NODES * H, 256), 256, 0, stream>>>(nmax, 0u, (size_t)N_NODES * H);
    fill_u32_k<<<cdivu((size_t)N_NODES * H, 256), 256, 0, stream>>>((unsigned*)nsum, 0u, (size_t)N_NODES * H);
    fill_u32_k<<<cdivu((size_t)N_NODES * D, 256), 256, 0, stream>>>((unsigned*)outacc, 0u, (size_t)N_NODES * D);

    edge_alpha_k<<<8192, 256, 0, stream>>>(
        xl, xr, embf, esrc, edst, att_w + (size_t)l * H * C, logit, nmax);
    alpha_exp_k<<<cdivu((size_t)EF * H, 256), 256, 0, stream>>>(logit, nmax, nsum, edst);
    aggregate_k<<<16384, 256, 0, stream>>>(logit, nsum, xl, esrc, edst, outacc);

    layer_post_k<<<N_NODES, 256, 0, stream>>>(
        outacc, conv_b + (size_t)l * D, ln_g + (size_t)l * D, ln_b + (size_t)l * D, h, hbf);
  }

  // ---- pooling + MLP ----
  fill_u32_k<<<cdivu((size_t)BGR * D, 256), 256, 0, stream>>>((unsigned*)xs, 0u, (size_t)BGR * D);
  fill_u32_k<<<cdivu((size_t)BGR, 256), 256, 0, stream>>>((unsigned*)gcnt, 0u, (size_t)BGR);
  pool_accum_k<<<N_NODES, 256, 0, stream>>>(h, batch, xs, gcnt);
  build_xg_k<<<BGR, 512, 0, stream>>>(xs, gcnt, xg, xgbf);

  transpose_bf16_k<<<cdivu((size_t)2 * D * D, 256), 256, 0, stream>>>(p1_w, p1T, 2 * D, D);
  transpose_bf16_k<<<cdivu((size_t)D * P, 256), 256, 0, stream>>>(p2_w, p2T, D, P);

  dim3 g1(BGR / 32, 1);
  wmma_gemm_bf16_k<<<g1, 256, 0, stream>>>(xgbf, p1T, p1_b, z, nullptr, BGR, 2 * D, D);
  post_mlp1_k<<<BGR, 256, 0, stream>>>(z, pln1_g, pln1_b, zbf);

  dim3 g2(BGR / 32, P / 256);
  wmma_gemm_bf16_k<<<g2, 256, 0, stream>>>(zbf, p2T, p2_b, opre, nullptr, BGR, D, P);
  final_ln_k<<<BGR, 512, 0, stream>>>(opre, pln2_g, pln2_b, (float*)d_out);
}
